// ComplexApproximateBNN_43198781063324
// MI455X (gfx1250) — matrix operand order repaired and checked by reading, not compiled, not run on hardware
//
#include <hip/hip_runtime.h>
#include <hip/hip_bf16.h>

typedef unsigned short ushort_t;
typedef __attribute__((ext_vector_type(16))) __bf16 v16bf;
typedef __attribute__((ext_vector_type(8)))  float  v8f;
typedef __attribute__((ext_vector_type(4)))  unsigned v4u;
typedef __attribute__((ext_vector_type(8)))  int    v8i;
typedef __attribute__((ext_vector_type(4)))  int    v4i;

// Problem constants (match reference)
constexpr int Bc = 128, Tc = 512, INc = 256, Xc = 512, OUTc = 128;
constexpr int BT = Bc * Tc;
constexpr int SCAN_BLOCKS = 64;   // 8 m-bands x 8 n-groups

// ---------------- helpers ----------------

__device__ __forceinline__ ushort_t f2bf(float f) {
    unsigned u = __float_as_uint(f);
    unsigned r = 0x7fffu + ((u >> 16) & 1u);      // round-to-nearest-even
    return (ushort_t)((u + r) >> 16);
}

__device__ __forceinline__ float actf(float h, int idx) {
    switch (idx) {
    case 0: return fmaxf(h, 0.0f);                          // ReLU
    case 1: return 1.0f / (1.0f + __expf(-h));              // Sigmoid
    case 2: return tanhf(h);                                // Tanh
    case 3: return h > 0.0f ? h : 0.1f * h;                 // LeakyReLU(0.1)
    default: {                                              // SELU
        const float alpha = 1.6732632423543772f;
        const float scale = 1.0507009873554805f;
        return scale * (h > 0.0f ? h : alpha * (__expf(h) - 1.0f));
    }
    }
}

// A fragment: 16x32 bf16, row-major A (M x K), stride lda (global memory).
__device__ __forceinline__ v16bf ld_a_frag(const ushort_t* A, int lda, int mBase, int kBase) {
    int lane = threadIdx.x & 31;
    int l = lane & 15;
    int hi8 = (lane >> 4) & 1;
    const ushort_t* p = A + (size_t)(mBase + l) * lda + kBase + hi8 * 8;
    union { uint4 u[2]; v16bf v; } u;
    u.u[0] = *(const uint4*)p;           // K group +0
    u.u[1] = *(const uint4*)(p + 16);    // K group +16
    return u.v;
}

// A fragment from a 16 x Xc LDS band (row stride Xc) -> ds_load_b128
__device__ __forceinline__ v16bf ld_a_frag_lds(const ushort_t* sA, int kBase) {
    int lane = threadIdx.x & 31;
    int l = lane & 15;
    int hi8 = (lane >> 4) & 1;
    const ushort_t* p = sA + l * Xc + kBase + hi8 * 8;
    union { uint4 u[2]; v16bf v; } u;
    u.u[0] = *(const uint4*)p;
    u.u[1] = *(const uint4*)(p + 16);
    return u.v;
}

// B fragment: 32x16 bf16 where B[k][n] = W[n][k], W row-major (N x K), stride ldw.
__device__ __forceinline__ v16bf ld_b_frag(const ushort_t* W, int ldw, int nBase, int kBase) {
    int lane = threadIdx.x & 31;
    int l = lane & 15;
    int hi16 = (lane >> 4) & 1;
    const ushort_t* p = W + (size_t)(nBase + l) * ldw + kBase + hi16 * 8;
    union { uint4 u[2]; v16bf v; } u;
    u.u[0] = *(const uint4*)p;
    u.u[1] = *(const uint4*)(p + 16);
    return u.v;
}

__device__ __forceinline__ v8f wmma_bf16(v16bf a, v16bf b, v8f c) {
    return __builtin_amdgcn_wmma_f32_16x16x32_bf16(
        false, a, false, b, (short)0, c, false, false);
}

// ---- Tensor Data Mover: stage 16 contiguous rows of 1024B (16x512 bf16) into LDS ----
// D# per CDNA5 ISA: group0 = {count/lds_addr/global_addr/type},
// group1 = {data_size, tensor dims, tile dims, dim0 stride}; 2-D tile -> groups 2/3 zero.
__device__ __forceinline__ void tdm_stage_rows16(unsigned ldsOff, const ushort_t* g) {
    unsigned long long ga = (unsigned long long)(size_t)g;
    const unsigned u0 = (Xc * 2) / 8;     // 128 units of 8B per row
    const unsigned rows = 16;
    v4u g0;
    g0.x = 1u;                                                 // count=1, user mode
    g0.y = ldsOff;                                             // lds_addr (bytes)
    g0.z = (unsigned)ga;                                       // global_addr[31:0]
    g0.w = (unsigned)((ga >> 32) & 0x1FFFFFFu) | (2u << 30);   // global_addr[56:32] | type=2
    v8i g1;
    g1[0] = (int)(3u << 16);                                   // data_size = 8 bytes
    g1[1] = (int)((u0 & 0xffffu) << 16);                       // tensor_dim0[15:0]
    g1[2] = (int)(((u0 >> 16) & 0xffffu) | (rows << 16));      // tensor_dim0 hi | tensor_dim1 lo
    g1[3] = (int)((u0 & 0xffffu) << 16);                       // tensor_dim1 hi(0) | tile_dim0
    g1[4] = (int)rows;                                         // tile_dim1 ; tile_dim2=0
    g1[5] = (int)u0;                                           // tensor_dim0_stride[31:0] (contiguous)
    g1[6] = 0;                                                 // stride hi | dim1 stride lo
    g1[7] = 0;
    v4i gz = {};
#if __has_include(<hip/amd_detail/amd_gfx1250_TDM.h>)
    v8i gz8 = {};
    __builtin_amdgcn_tensor_load_to_lds(g0, g1, gz, gz, gz8, 0);
#else
    __builtin_amdgcn_tensor_load_to_lds(g0, g1, gz, gz, 0);
#endif
}

__device__ __forceinline__ void stage_band(ushort_t* sDst, const ushort_t* gSrc) {
    if ((threadIdx.x >> 5) == 0) {
        tdm_stage_rows16((unsigned)(size_t)sDst, gSrc);
        __builtin_amdgcn_s_wait_tensorcnt(0);
    }
    __syncthreads();
}

__device__ __forceinline__ void stage_band2(ushort_t* s0, const ushort_t* gA,
                                            ushort_t* s1, const ushort_t* gB) {
    if ((threadIdx.x >> 5) == 0) {
        tdm_stage_rows16((unsigned)(size_t)s0, gA);
        tdm_stage_rows16((unsigned)(size_t)s1, gB);
        __builtin_amdgcn_s_wait_tensorcnt(0);
    }
    __syncthreads();
}

// C tile epilogue: VGPR v, lanes 0-15 -> (M=v, N=lane); lanes 16-31 -> (M=v+8, N=lane-16)
__device__ __forceinline__ void epilogue_bf16(const v8f& c, const float* add /*ld=Xc or null*/,
                                              const float* bias, const int* aidx,
                                              ushort_t* out, int ldo, int mB, int nB) {
    int lane = threadIdx.x & 31;
    int n = nB + (lane & 15);
    int mAdd = (lane >> 4) * 8;
    float bi = bias ? bias[n] : 0.0f;
    int ai = aidx[n];
#pragma unroll
    for (int v = 0; v < 8; ++v) {
        int m = mB + v + mAdd;
        float h = c[v] + bi;
        if (add) h += add[(size_t)m * Xc + n];
        out[(size_t)m * ldo + n] = f2bf(actf(h, ai));
    }
}

// ---------------- kernels ----------------

__global__ void cvt_f2bf_kernel(const float* __restrict__ in, ushort_t* __restrict__ out, int n) {
    int i = blockIdx.x * blockDim.x + threadIdx.x;
    int s = gridDim.x * blockDim.x;
    for (; i < n; i += s) out[i] = f2bf(in[i]);
}

// Parallel precompute GEMMs over all B*T rows. One 16x16 tile per wave iteration.
// mode 0: out_bf16[m*N+n] = act(c + bias[n], aidx[n])          (h0)
// mode 1: out_f32[((m%T)*B + m/T)*N + n] = c + bias[n]         (pre1, reordered (t,b))
__global__ __launch_bounds__(256) void gemm_pre_kernel(
    const ushort_t* __restrict__ A, int lda,
    const ushort_t* __restrict__ W, int ldw, int kLen,
    const float* __restrict__ bias, const int* __restrict__ aidx,
    int M, int N, ushort_t* __restrict__ outB, float* __restrict__ outF, int mode)
{
    int gw = blockIdx.x * (blockDim.x >> 5) + (threadIdx.x >> 5);
    int nW = gridDim.x * (blockDim.x >> 5);
    int mTiles = M >> 4;
    int nTiles = mTiles * (N >> 4);
    int lane = threadIdx.x & 31;
    int nl = lane & 15;
    int mAdd = (lane >> 4) * 8;

    for (int tile = gw; tile < nTiles; tile += nW) {
        int mB = (tile % mTiles) * 16;
        int nB = (tile / mTiles) * 16;
        v8f c = {};
        for (int k = 0; k < kLen; k += 32) {
            v16bf a = ld_a_frag(A, lda, mB, k);
            v16bf b = ld_b_frag(W, ldw, nB, k);
            c = wmma_bf16(a, b, c);
        }
        int n = nB + nl;
        float bi = bias[n];
        if (mode == 0) {
            int ai = aidx[n];
#pragma unroll
            for (int v = 0; v < 8; ++v) {
                int m = mB + v + mAdd;
                outB[(size_t)m * N + n] = f2bf(actf(c[v] + bi, ai));
            }
        } else {
#pragma unroll
            for (int v = 0; v < 8; ++v) {
                int m = mB + v + mAdd;
                int tt = m & (Tc - 1);
                int bb = m >> 9;           // Tc == 512
                outF[((size_t)tt * Bc + bb) * N + n] = c[v] + bi;
            }
        }
    }
}

__device__ __forceinline__ void grid_barrier(unsigned* ctr, unsigned& gen, int nblocks) {
    __threadfence();           // make this block's stores device-visible
    __syncthreads();
    gen += (unsigned)nblocks;
    if (threadIdx.x == 0) {
        __hip_atomic_fetch_add(ctr, 1u, __ATOMIC_RELEASE, __HIP_MEMORY_SCOPE_AGENT);
        while (__hip_atomic_load(ctr, __ATOMIC_ACQUIRE, __HIP_MEMORY_SCOPE_AGENT) < gen)
            __builtin_amdgcn_s_sleep(2);
    }
    __syncthreads();
    __threadfence();
}

// Persistent sequential-scan kernel: 512 steps, 6 grid-barriered sub-phases each.
// Block b owns m-band (b & 7) of 16 batch rows; its A band is TDM-staged in LDS.
__global__ __launch_bounds__(256) void scan_kernel(
    const float* __restrict__ pre1,          // (T, B, X) f32, includes b_h0
    const ushort_t* __restrict__ wH0b,       // (X, 2X) bf16 (use cols X..2X-1)
    const ushort_t* __restrict__ wH1b,       // (X, X)
    const ushort_t* __restrict__ wH2b,       // (X, 2X)
    const ushort_t* __restrict__ wH3b,       // (X, 2X)
    const ushort_t* __restrict__ wRecb,      // (X, X)
    const ushort_t* __restrict__ wOutb,      // (OUT, X)
    const float* __restrict__ b_h1, const float* __restrict__ b_h2,
    const float* __restrict__ b_h3, const float* __restrict__ b_rec,
    const float* __restrict__ b_out,
    const int* __restrict__ actIdx,          // (5, X)
    const int* __restrict__ recIdx,          // (X,)
    const int* __restrict__ outIdx,          // (OUT,)
    ushort_t* __restrict__ rnb,              // (B, X) bf16
    ushort_t* __restrict__ h1b, ushort_t* __restrict__ h2b,
    ushort_t* __restrict__ h3b, ushort_t* __restrict__ h4b,
    float* __restrict__ r,                   // (B, X) f32 (zeroed before launch)
    float* __restrict__ out,                 // (B, T, OUT) f32
    unsigned* __restrict__ barCtr)           // zeroed before launch
{
    __shared__ ushort_t sA[2][16 * Xc];      // two 16x512 bf16 bands (2 x 16KB)

    const int wid = threadIdx.x >> 5;
    const int gw = blockIdx.x * (blockDim.x >> 5) + wid;
    const int nW = gridDim.x * (blockDim.x >> 5);
    const int lane = threadIdx.x & 31;
    const int mB = (blockIdx.x & 7) * 16;    // this block's m-band
    const int nt = (blockIdx.x >> 3) * 8 + wid;   // this wave's n-tile (0..63)
    unsigned gen = 0;

    for (int t = 0; t < Tc; ++t) {
        // ---- Phase A: rn = l2norm(r), stored bf16 ----
        for (int row = gw; row < Bc; row += nW) {
            const float* rp = r + (size_t)row * Xc;
            float s = 0.0f;
            for (int cidx = lane; cidx < Xc; cidx += 32) { float v = rp[cidx]; s += v * v; }
#pragma unroll
            for (int off = 16; off > 0; off >>= 1) s += __shfl_xor(s, off);
            float scale = 1.0f / fmaxf(sqrtf(s), 1e-12f);
            ushort_t* op = rnb + (size_t)row * Xc;
            for (int cidx = lane; cidx < Xc; cidx += 32) op[cidx] = f2bf(rp[cidx] * scale);
        }
        grid_barrier(barCtr, gen, gridDim.x);

        // ---- Phase B: h1 = act1(pre1[t] + rn @ W_h0[:,X:]ᵀ) ----
        const float* pre1t = pre1 + (size_t)t * Bc * Xc;
        stage_band(sA[0], rnb + (size_t)mB * Xc);
        if (nt < 32) {
            int nB = nt * 16;
            v8f c = {};
            for (int k = 0; k < Xc; k += 32) {
                v16bf a = ld_a_frag_lds(sA[0], k);
                v16bf b = ld_b_frag(wH0b, 2 * Xc, nB, Xc + k);
                c = wmma_bf16(a, b, c);
            }
            epilogue_bf16(c, pre1t, nullptr, actIdx + 1 * Xc, h1b, Xc, mB, nB);
        }
        grid_barrier(barCtr, gen, gridDim.x);

        // ---- Phase C: h2 = act2(h1 @ W_h1ᵀ + b_h1) ----
        stage_band(sA[0], h1b + (size_t)mB * Xc);
        if (nt < 32) {
            int nB = nt * 16;
            v8f c = {};
            for (int k = 0; k < Xc; k += 32) {
                v16bf a = ld_a_frag_lds(sA[0], k);
                v16bf b = ld_b_frag(wH1b, Xc, nB, k);
                c = wmma_bf16(a, b, c);
            }
            epilogue_bf16(c, nullptr, b_h1, actIdx + 2 * Xc, h2b, Xc, mB, nB);
        }
        grid_barrier(barCtr, gen, gridDim.x);

        // ---- Phase D: h3 = act3([h2, h1] @ W_h2ᵀ + b_h2) ----
        stage_band2(sA[0], h2b + (size_t)mB * Xc, sA[1], h1b + (size_t)mB * Xc);
        if (nt < 32) {
            int nB = nt * 16;
            v8f c = {};
            for (int k = 0; k < Xc; k += 32) {
                v16bf a = ld_a_frag_lds(sA[0], k);
                v16bf b = ld_b_frag(wH2b, 2 * Xc, nB, k);
                c = wmma_bf16(a, b, c);
            }
            for (int k = 0; k < Xc; k += 32) {
                v16bf a = ld_a_frag_lds(sA[1], k);
                v16bf b = ld_b_frag(wH2b, 2 * Xc, nB, Xc + k);
                c = wmma_bf16(a, b, c);
            }
            epilogue_bf16(c, nullptr, b_h2, actIdx + 3 * Xc, h3b, Xc, mB, nB);
        }
        grid_barrier(barCtr, gen, gridDim.x);

        // ---- Phase E: h4 = act4([h3, h2] @ W_h3ᵀ + b_h3) ----
        stage_band2(sA[0], h3b + (size_t)mB * Xc, sA[1], h2b + (size_t)mB * Xc);
        if (nt < 32) {
            int nB = nt * 16;
            v8f c = {};
            for (int k = 0; k < Xc; k += 32) {
                v16bf a = ld_a_frag_lds(sA[0], k);
                v16bf b = ld_b_frag(wH3b, 2 * Xc, nB, k);
                c = wmma_bf16(a, b, c);
            }
            for (int k = 0; k < Xc; k += 32) {
                v16bf a = ld_a_frag_lds(sA[1], k);
                v16bf b = ld_b_frag(wH3b, 2 * Xc, nB, Xc + k);
                c = wmma_bf16(a, b, c);
            }
            epilogue_bf16(c, nullptr, b_h3, actIdx + 4 * Xc, h4b, Xc, mB, nB);
        }
        grid_barrier(barCtr, gen, gridDim.x);

        // ---- Phase F: r_new = act_rec(h4 @ W_recᵀ + b_rec); y = act_out(h4 @ W_outᵀ + b_out) ----
        stage_band(sA[0], h4b + (size_t)mB * Xc);
        if (nt < 32) {
            int nB = nt * 16;
            v8f c = {};
            for (int k = 0; k < Xc; k += 32) {
                v16bf a = ld_a_frag_lds(sA[0], k);
                v16bf b = ld_b_frag(wRecb, Xc, nB, k);
                c = wmma_bf16(a, b, c);
            }
            int n = nB + (lane & 15);
            int mAdd = (lane >> 4) * 8;
            float bi = b_rec[n];
            int ai = recIdx[n];
#pragma unroll
            for (int v = 0; v < 8; ++v) {
                int m = mB + v + mAdd;
                r[(size_t)m * Xc + n] = actf(c[v] + bi, ai);
            }
        } else if (nt < 40) {
            int nB = (nt - 32) * 16;
            v8f c = {};
            for (int k = 0; k < Xc; k += 32) {
                v16bf a = ld_a_frag_lds(sA[0], k);
                v16bf b = ld_b_frag(wOutb, Xc, nB, k);
                c = wmma_bf16(a, b, c);
            }
            int n = nB + (lane & 15);
            int mAdd = (lane >> 4) * 8;
            float bi = b_out[n];
            int ai = outIdx[n];
#pragma unroll
            for (int v = 0; v < 8; ++v) {
                int m = mB + v + mAdd;   // m = batch index b
                out[((size_t)m * Tc + t) * OUTc + n] = actf(c[v] + bi, ai);
            }
        }
        grid_barrier(barCtr, gen, gridDim.x);   // r visible before next step's l2norm
    }
}

// ---------------- host launcher ----------------

extern "C" void kernel_launch(void* const* d_in, const int* in_sizes, int n_in,
                              void* d_out, int out_size, void* d_ws, size_t ws_size,
                              hipStream_t stream) {
    (void)in_sizes; (void)n_in; (void)out_size; (void)ws_size;

    const float* x     = (const float*)d_in[0];
    const float* W_in  = (const float*)d_in[1];
    const float* b_in  = (const float*)d_in[2];
    const float* W_h0  = (const float*)d_in[3];
    const float* b_h0  = (const float*)d_in[4];
    const float* W_h1  = (const float*)d_in[5];
    const float* b_h1  = (const float*)d_in[6];
    const float* W_h2  = (const float*)d_in[7];
    const float* b_h2  = (const float*)d_in[8];
    const float* W_h3  = (const float*)d_in[9];
    const float* b_h3  = (const float*)d_in[10];
    const float* W_rec = (const float*)d_in[11];
    const float* b_rec = (const float*)d_in[12];
    const float* W_out = (const float*)d_in[13];
    const float* b_out = (const float*)d_in[14];
    const int* act_idx = (const int*)d_in[15];
    const int* rec_idx = (const int*)d_in[16];
    const int* out_idx = (const int*)d_in[17];
    float* out = (float*)d_out;

    // workspace carve-up (256B aligned)
    char* base = (char*)d_ws;
    size_t off = 0;
    auto carve = [&](size_t bytes) -> char* {
        char* p = base + off;
        off = (off + bytes + 255) & ~(size_t)255;
        return p;
    };
    unsigned*  barCtr = (unsigned*) carve(256);
    float*     rbuf   = (float*)    carve((size_t)Bc * Xc * 4);
    ushort_t*  rnb    = (ushort_t*) carve((size_t)Bc * Xc * 2);
    ushort_t*  h1b    = (ushort_t*) carve((size_t)Bc * Xc * 2);
    ushort_t*  h2b    = (ushort_t*) carve((size_t)Bc * Xc * 2);
    ushort_t*  h3b    = (ushort_t*) carve((size_t)Bc * Xc * 2);
    ushort_t*  h4b    = (ushort_t*) carve((size_t)Bc * Xc * 2);
    ushort_t*  wInb   = (ushort_t*) carve((size_t)Xc * INc * 2);
    ushort_t*  wH0b   = (ushort_t*) carve((size_t)Xc * 2 * Xc * 2);
    ushort_t*  wH1b   = (ushort_t*) carve((size_t)Xc * Xc * 2);
    ushort_t*  wH2b   = (ushort_t*) carve((size_t)Xc * 2 * Xc * 2);
    ushort_t*  wH3b   = (ushort_t*) carve((size_t)Xc * 2 * Xc * 2);
    ushort_t*  wRecb  = (ushort_t*) carve((size_t)Xc * Xc * 2);
    ushort_t*  wOutb  = (ushort_t*) carve((size_t)OUTc * Xc * 2);
    ushort_t*  xb     = (ushort_t*) carve((size_t)BT * INc * 2);
    ushort_t*  h0b    = (ushort_t*) carve((size_t)BT * Xc * 2);
    float*     pre1   = (float*)    carve((size_t)BT * Xc * 4);

    hipMemsetAsync(barCtr, 0, 256, stream);
    hipMemsetAsync(rbuf, 0, (size_t)Bc * Xc * 4, stream);   // r0 = zeros

    // fp32 -> bf16 conversions
    cvt_f2bf_kernel<<<2048, 256, 0, stream>>>(x, xb, BT * INc);
    cvt_f2bf_kernel<<<256, 256, 0, stream>>>(W_in,  wInb,  Xc * INc);
    cvt_f2bf_kernel<<<256, 256, 0, stream>>>(W_h0,  wH0b,  Xc * 2 * Xc);
    cvt_f2bf_kernel<<<256, 256, 0, stream>>>(W_h1,  wH1b,  Xc * Xc);
    cvt_f2bf_kernel<<<256, 256, 0, stream>>>(W_h2,  wH2b,  Xc * 2 * Xc);
    cvt_f2bf_kernel<<<256, 256, 0, stream>>>(W_h3,  wH3b,  Xc * 2 * Xc);
    cvt_f2bf_kernel<<<256, 256, 0, stream>>>(W_rec, wRecb, Xc * Xc);
    cvt_f2bf_kernel<<<256, 256, 0, stream>>>(W_out, wOutb, OUTc * Xc);

    // Parallel precompute across all B*T rows:
    // h0 = act0(x @ W_inᵀ + b_in)           (BT x X, bf16)
    gemm_pre_kernel<<<4096, 256, 0, stream>>>(xb, INc, wInb, INc, INc,
                                              b_in, act_idx, BT, Xc, h0b, nullptr, 0);
    // pre1 = h0 @ W_h0[:, :X]ᵀ + b_h0       ((T,B,X) f32)
    gemm_pre_kernel<<<4096, 256, 0, stream>>>(h0b, Xc, wH0b, 2 * Xc, Xc,
                                              b_h0, act_idx, BT, Xc, nullptr, pre1, 1);

    // Sequential scan (persistent, grid-barriered, TDM-staged A bands)
    scan_kernel<<<SCAN_BLOCKS, 256, 0, stream>>>(
        pre1, wH0b, wH1b, wH2b, wH3b, wRecb, wOutb,
        b_h1, b_h2, b_h3, b_rec, b_out,
        act_idx, rec_idx, out_idx,
        rnb, h1b, h2b, h3b, h4b, rbuf, out, barCtr);
}
